// HeteroGraphAttentionEmbedding_52819507806812
// MI455X (gfx1250) — matrix operand order repaired and checked by reading, not compiled, not run on hardware
//
#include <hip/hip_runtime.h>

// ---------------- problem constants (from reference) ----------------
#define N_A   100000
#define N_B   100000
#define N_TOT 200000
#define NEDGE 400000
#define IN_C  128
#define F_DIM 128
#define H_NUM 2
#define D_DIM 64
#define TD_N  100
#define E_DIM 164          // MSG + TD
#define E_PAD 192          // K padded to multiple of 32 for WMMA
#define W3F   384          // 3 * F

typedef __bf16 bf16_t;
typedef __attribute__((ext_vector_type(16))) __bf16 bf16x16;
typedef __attribute__((ext_vector_type(8)))  __bf16 bf16x8;
typedef __attribute__((ext_vector_type(8)))  float  f32x8;

__device__ __forceinline__ f32x8 wmma_bf16(bf16x16 a, bf16x16 b, f32x8 c) {
  // (neg_a, A, neg_b, B, c_mod, C, reuse_a, reuse_b) -> v_wmma_f32_16x16x32_bf16
  return __builtin_amdgcn_wmma_f32_16x16x32_bf16(false, a, false, b, (short)0, c,
                                                 false, false);
}

__device__ __forceinline__ float gelu_exact(float x) {
  return 0.5f * x * (1.0f + erff(x * 0.70710678118654752f));
}

__device__ __forceinline__ void atomicMaxF32(float* addr, float val) {
  unsigned int* ua = (unsigned int*)addr;
  unsigned int cur = __float_as_uint(*addr);
  while (__uint_as_float(cur) < val) {
    unsigned int prev = atomicCAS(ua, cur, __float_as_uint(val));
    if (prev == cur) break;
    cur = prev;
  }
}

// ---------------- small prep kernels ----------------

// Fold the per-head relation transform (HeteroLinear) into the kqv projection:
//   k_src = (x@Wk + bk) @ Wrel[h,et]  ==  x @ (Wk@Wrel) + bk@Wrel
// Output layout [128][384]: cols 0..127 = composed-k, 128..255 = q (copy),
// 256..383 = composed-v. i==128 encodes the bias row.
__global__ void compose_kernel(const float* __restrict__ WA, const float* __restrict__ bA,
                               const float* __restrict__ WB, const float* __restrict__ bB,
                               const float* __restrict__ Wk_rel, const float* __restrict__ Wv_rel,
                               float* __restrict__ WcA, float* __restrict__ bcA,
                               float* __restrict__ WcB, float* __restrict__ bcB) {
  int t = blockIdx.x * blockDim.x + threadIdx.x;
  const int TOT = 2 * W3F * 129;
  if (t >= TOT) return;
  int type = t / (W3F * 129);
  int rem  = t % (W3F * 129);
  int col  = rem / 129;
  int i    = rem % 129;            // 128 -> bias row
  const float* W = type ? WB : WA;
  const float* b = type ? bB : bA;
  float* Wc = type ? WcB : WcA;
  float* bc = type ? bcB : bcA;
  float val;
  if (col >= 128 && col < 256) {   // q block: identity
    val = (i < 128) ? W[i * W3F + col] : b[col];
  } else {
    const bool isK = (col < 128);
    const int cc = isK ? col : col - 256;
    const int h = cc >> 6, eo = cc & 63;
    // relation type index = h*NE + et ; et==type (A->B uses 0, B->A uses 1)
    const float* R = (isK ? Wk_rel : Wv_rel) + (size_t)(h * 2 + type) * D_DIM * D_DIM;
    const int srcbase = (isK ? 0 : 256) + h * 64;
    float s = 0.f;
    if (i < 128) {
      for (int d = 0; d < 64; ++d) s += W[i * W3F + srcbase + d] * R[d * 64 + eo];
    } else {
      for (int d = 0; d < 64; ++d) s += b[srcbase + d] * R[d * 64 + eo];
    }
    val = s;
  }
  if (i < 128) Wc[i * W3F + col] = val; else bc[col] = val;
}

// Pack a row-major f32 matrix W[K][N] into WMMA B-fragment-major bf16:
//   P[((nt*KB + kb)*32 + lane)*16 + j]  with  Kidx = kb*32 + (j<8?0:16) + (lane>=16?8:0) + j%8,
//   col = nt*16 + (lane&15).  Rows >= Kin are zero (K padding).
// One thread per 16-element lane fragment.
__global__ void pack_b_kernel(const float* __restrict__ W, bf16_t* __restrict__ P,
                              int K, int N, int Kin) {
  const int t = blockIdx.x * blockDim.x + threadIdx.x;
  const int KB = K / 32;
  const int total = (N / 16) * KB * 32;
  if (t >= total) return;
  const int nt   = t / (KB * 32);
  const int rem  = t % (KB * 32);
  const int kb   = rem / 32;
  const int lane = rem % 32;
  const int m = lane & 15, hi = lane >> 4;
  const int col = nt * 16 + m;
  bf16x16 b;
  #pragma unroll
  for (int j = 0; j < 16; ++j) {
    const int kidx = kb * 32 + (j < 8 ? 0 : 16) + hi * 8 + (j & 7);
    b[j] = (kidx < Kin) ? (bf16_t)W[(size_t)kidx * N + col] : (bf16_t)0.0f;
  }
  ((bf16x16*)P)[t] = b;
}

// Pack a row-major f32 matrix X[Nrows][Kc] into WMMA A-fragment-major bf16:
//   element j of lane fragment: M = lane&15, K = kb*32 + (j<8?0:16) + (lane>=16?8:0) + j%8
__global__ void pack_a_kernel(const float* __restrict__ X, bf16_t* __restrict__ P,
                              int Nrows, int Kc) {
  const int t = blockIdx.x * blockDim.x + threadIdx.x;
  const int KB = Kc / 32;
  const int total = (Nrows / 16) * KB * 32;
  if (t >= total) return;
  const int rt   = t / (KB * 32);
  const int rem  = t % (KB * 32);
  const int kb   = rem / 32;
  const int lane = rem % 32;
  const int m = lane & 15, hi = lane >> 4;
  const float* xr = X + (size_t)(rt * 16 + m) * Kc + kb * 32 + hi * 8;
  bf16x16 a;
  #pragma unroll
  for (int j = 0; j < 8; ++j) a[j]     = (bf16_t)xr[j];
  #pragma unroll
  for (int j = 0; j < 8; ++j) a[8 + j] = (bf16_t)xr[16 + j];
  ((bf16x16*)P)[t] = a;
}

__global__ void init_neg_kernel(float* __restrict__ m, int n) {
  int t = blockIdx.x * blockDim.x + threadIdx.x;
  if (t < n) m[t] = -3.0e38f;
}

// ---------------- kqv GEMM: OUT[N,384] = X[N,128]@Wc + bc (packed WMMA bf16) ----------------
__global__ __launch_bounds__(32)
void gemm_kqv_kernel(const bf16_t* __restrict__ PA, const bf16_t* __restrict__ PB,
                     const float* __restrict__ bc, float* __restrict__ OUT) {
  const int lane = threadIdx.x;
  const int m = lane & 15, hi = lane >> 4;
  const int rt = blockIdx.x;          // 16-row tile
  const int cg = blockIdx.y;          // 64-column group (0..5)
  const bf16x16* pa = (const bf16x16*)PA;
  const bf16x16* pb = (const bf16x16*)PB;
  f32x8 acc[4] = {};
  #pragma unroll
  for (int kb = 0; kb < 4; ++kb) {                       // K = 128
    const bf16x16 a = pa[(size_t)(rt * 4 + kb) * 32 + lane];
    #pragma unroll
    for (int ct = 0; ct < 4; ++ct) {
      const int nt = cg * 4 + ct;
      const bf16x16 b = pb[(size_t)(nt * 4 + kb) * 32 + lane];
      acc[ct] = wmma_bf16(a, b, acc[ct]);
    }
  }
  #pragma unroll
  for (int ct = 0; ct < 4; ++ct) {
    const int col = cg * 64 + ct * 16 + m;
    const float bv = bc[col];
    #pragma unroll
    for (int r = 0; r < 8; ++r) {
      const int row = rt * 16 + r + hi * 8;
      OUT[(size_t)row * W3F + col] = acc[ct][r] + bv;
    }
  }
}

// ---------------- edge features: [cos(dt*w+b) || msg] @ W_edge + b_edge ----------------
__global__ __launch_bounds__(32)
void edge_feat_kernel(const int* __restrict__ idxAB, const int* __restrict__ idxBA,
                      const float* __restrict__ luA, const float* __restrict__ luB,
                      const float* __restrict__ tAB, const float* __restrict__ tBA,
                      const float* __restrict__ msgAB, const float* __restrict__ msgBA,
                      const float* __restrict__ time_w, const float* __restrict__ time_b,
                      const bf16_t* __restrict__ PWedge, const float* __restrict__ b_edge,
                      float* __restrict__ EF) {
  __shared__ __attribute__((aligned(32))) bf16_t efLds[16][E_PAD];  // 6 KB of 320 KB WGP LDS
  const int lane = threadIdx.x;
  const int row = lane & 15;
  const int hi  = lane >> 4;
  const long long e0 = (long long)blockIdx.x * 16;   // E multiple of 16 -> no mixed tiles
  const bool isAB = (e0 < NEDGE);
  const long long el = isAB ? (e0 + row) : (e0 + row - NEDGE);
  const int src = (isAB ? idxAB : idxBA)[el];
  const float rt = (isAB ? luA : luB)[src] - (isAB ? tAB : tBA)[el];
  const float* msg = (isAB ? msgAB : msgBA) + el * 64;
  const int c0 = hi * 96;
  for (int c = c0; c < c0 + 96; ++c) {
    float v;
    if (c < TD_N)       v = cosf(rt * time_w[c] + time_b[c]);
    else if (c < E_DIM) v = msg[c - TD_N];
    else                v = 0.0f;
    efLds[row][c] = (bf16_t)v;
  }
  __syncthreads();
  const bf16x16* pb = (const bf16x16*)PWedge;
  f32x8 acc[4] = {};
  #pragma unroll
  for (int kb = 0; kb < 6; ++kb) {                       // K = 192
    const int kk = kb * 32;
    const bf16x8 alo = *(const bf16x8*)&efLds[row][kk + hi * 8];        // ds_load_b128
    const bf16x8 ahi = *(const bf16x8*)&efLds[row][kk + 16 + hi * 8];   // ds_load_b128
    bf16x16 a;
    #pragma unroll
    for (int j = 0; j < 8; ++j) { a[j] = alo[j]; a[8 + j] = ahi[j]; }
    #pragma unroll
    for (int ct = 0; ct < 4; ++ct) {
      const bf16x16 b = pb[(size_t)(ct * 6 + kb) * 32 + lane];
      acc[ct] = wmma_bf16(a, b, acc[ct]);
    }
  }
  #pragma unroll
  for (int ct = 0; ct < 4; ++ct) {
    const int col = ct * 16 + row;
    const float bv = b_edge[col];
    #pragma unroll
    for (int r = 0; r < 8; ++r) {
      const long long ee = e0 + r + hi * 8;
      EF[ee * 64 + col] = acc[ct][r] + bv;
    }
  }
}

// ---------------- attention pass 1: alpha = (q_i . k_j) * p_rel * scale ----------------
__global__ void alpha_kernel(const int* __restrict__ idxAB, const int* __restrict__ idxBA,
                             const float* __restrict__ kqvA, const float* __restrict__ kqvB,
                             const float* __restrict__ prAB, const float* __restrict__ prBA,
                             float* __restrict__ alpha, float* __restrict__ mbuf) {
  const long long gid = (long long)blockIdx.x * blockDim.x + threadIdx.x;
  const long long e = gid >> 2;
  if (e >= 2LL * NEDGE) return;
  const int sub = (int)(gid & 3);             // 4 lanes per edge, 32 floats each
  const bool isAB = e < NEDGE;
  const long long el = isAB ? e : e - NEDGE;
  const int src  = (isAB ? idxAB : idxBA)[el];
  const int dstl = (isAB ? idxAB : idxBA)[NEDGE + el];
  const float* kp = (isAB ? kqvA : kqvB) + (size_t)src  * W3F + sub * 32;        // k_src cols 0..127
  const float* qp = (isAB ? kqvB : kqvA) + (size_t)dstl * W3F + 128 + sub * 32;  // q cols 128..255
  float dot = 0.f;
  #pragma unroll
  for (int i = 0; i < 8; ++i) {
    float4 q4 = ((const float4*)qp)[i];
    float4 k4 = ((const float4*)kp)[i];
    dot += q4.x * k4.x + q4.y * k4.y + q4.z * k4.z + q4.w * k4.w;
  }
  dot += __shfl_xor(dot, 1);                  // pair within same edge/head
  if ((sub & 1) == 0) {
    const int h = sub >> 1;
    const float pr = (isAB ? prAB : prBA)[h];
    const float a = dot * pr * 0.125f;        // scale = 1/sqrt(64)
    alpha[e * H_NUM + h] = a;
    const long long gdst = isAB ? (long long)N_A + dstl : (long long)dstl;
    atomicMaxF32(&mbuf[gdst * H_NUM + h], a);
  }
}

// ---------------- attention pass 2: e = exp(alpha - max), accumulate sums ----------------
__global__ void softmax_kernel(const int* __restrict__ idxAB, const int* __restrict__ idxBA,
                               float* __restrict__ alpha, const float* __restrict__ mbuf,
                               float* __restrict__ sbuf) {
  const long long gid = (long long)blockIdx.x * blockDim.x + threadIdx.x;
  if (gid >= 2LL * NEDGE * H_NUM) return;
  const long long e = gid >> 1;
  const int h = (int)(gid & 1);
  const bool isAB = e < NEDGE;
  const long long el = isAB ? e : e - NEDGE;
  const int dstl = (isAB ? idxAB : idxBA)[NEDGE + el];
  const long long gdst = isAB ? (long long)N_A + dstl : (long long)dstl;
  const float ex = expf(alpha[gid] - mbuf[gdst * H_NUM + h]);
  alpha[gid] = ex;
  atomicAdd(&sbuf[gdst * H_NUM + h], ex);
}

// ---------------- attention pass 3: scatter alpha * [v_j || ef] into agg ----------------
__global__ void scatter_kernel(const int* __restrict__ idxAB, const int* __restrict__ idxBA,
                               const float* __restrict__ kqvA, const float* __restrict__ kqvB,
                               const float* __restrict__ EF, const float* __restrict__ alpha,
                               const float* __restrict__ sbuf, float* __restrict__ agg) {
  const long long gid = (long long)blockIdx.x * blockDim.x + threadIdx.x;
  const long long e = gid >> 4;
  if (e >= 2LL * NEDGE) return;
  const int sub = (int)(gid & 15);            // 16 lanes/edge, 16 columns each of 256
  const bool isAB = e < NEDGE;
  const long long el = isAB ? e : e - NEDGE;
  const int src  = (isAB ? idxAB : idxBA)[el];
  const int dstl = (isAB ? idxAB : idxBA)[NEDGE + el];
  const long long gdst = isAB ? (long long)N_A + dstl : (long long)dstl;
  const int c0 = sub * 16;
  const int h = c0 >> 7;                      // column layout: [h=0: v|ef][h=1: v|ef]
  const int inner = c0 & 127;
  const float coef = alpha[e * H_NUM + h] / (sbuf[gdst * H_NUM + h] + 1e-16f);
  const float* srcv;
  if (inner < 64)
    srcv = (isAB ? kqvA : kqvB) + (size_t)src * W3F + 256 + h * 64 + inner;  // v_src cols 256..383
  else
    srcv = EF + e * 64 + (inner - 64);
  float* out = agg + gdst * 256 + c0;
  #pragma unroll
  for (int i = 0; i < 4; ++i) {
    float4 v = ((const float4*)srcv)[i];
    atomicAdd(out + i * 4 + 0, v.x * coef);
    atomicAdd(out + i * 4 + 1, v.y * coef);
    atomicAdd(out + i * 4 + 2, v.z * coef);
    atomicAdd(out + i * 4 + 3, v.w * coef);
  }
}

// ---------------- output: sigmoid-gated  gelu(agg)@W_out + b  vs  x ----------------
__global__ __launch_bounds__(32)
void out_proj_kernel(const float* __restrict__ AGG, const bf16_t* __restrict__ PWout,
                     const float* __restrict__ bout, const float* __restrict__ X,
                     const float* __restrict__ skip, float* __restrict__ OUT) {
  const int lane = threadIdx.x;
  const int m = lane & 15, hi = lane >> 4;
  const int rt = blockIdx.x;
  const int cg = blockIdx.y;          // 0..1
  const bf16x16* pb = (const bf16x16*)PWout;
  f32x8 acc[4] = {};
  #pragma unroll
  for (int kb = 0; kb < 8; ++kb) {                       // K = 256
    bf16x16 a;
    const float* ar = AGG + (size_t)(rt * 16 + m) * 256 + kb * 32 + hi * 8;
    #pragma unroll
    for (int j = 0; j < 8; ++j) a[j]     = (bf16_t)gelu_exact(ar[j]);
    #pragma unroll
    for (int j = 0; j < 8; ++j) a[8 + j] = (bf16_t)gelu_exact(ar[16 + j]);
    #pragma unroll
    for (int ct = 0; ct < 4; ++ct) {
      const int nt = cg * 4 + ct;
      const bf16x16 b = pb[(size_t)(nt * 8 + kb) * 32 + lane];
      acc[ct] = wmma_bf16(a, b, acc[ct]);
    }
  }
  const float g = 1.0f / (1.0f + expf(-skip[0]));
  #pragma unroll
  for (int ct = 0; ct < 4; ++ct) {
    const int col = cg * 64 + ct * 16 + m;
    const float bv = bout[col];
    #pragma unroll
    for (int r = 0; r < 8; ++r) {
      const int row = rt * 16 + r + hi * 8;
      const float o = acc[ct][r] + bv;
      OUT[(size_t)row * F_DIM + col] = g * o + (1.0f - g) * X[(size_t)row * F_DIM + col];
    }
  }
}

// ---------------- host launcher ----------------
extern "C" void kernel_launch(void* const* d_in, const int* in_sizes, int n_in,
                              void* d_out, int out_size, void* d_ws, size_t ws_size,
                              hipStream_t stream) {
  (void)in_sizes; (void)n_in; (void)out_size; (void)ws_size;
  const float* x_A    = (const float*)d_in[0];
  const float* x_B    = (const float*)d_in[1];
  const float* luA    = (const float*)d_in[2];
  const float* luB    = (const float*)d_in[3];
  const float* tAB    = (const float*)d_in[4];
  const float* tBA    = (const float*)d_in[5];
  const float* msgAB  = (const float*)d_in[6];
  const float* msgBA  = (const float*)d_in[7];
  const float* WkqvA  = (const float*)d_in[8];
  const float* bkqvA  = (const float*)d_in[9];
  const float* WkqvB  = (const float*)d_in[10];
  const float* bkqvB  = (const float*)d_in[11];
  const float* Wk_rel = (const float*)d_in[12];
  const float* Wv_rel = (const float*)d_in[13];
  const float* W_edge = (const float*)d_in[14];
  const float* b_edge = (const float*)d_in[15];
  const float* WoutA  = (const float*)d_in[16];
  const float* boutA  = (const float*)d_in[17];
  const float* WoutB  = (const float*)d_in[18];
  const float* boutB  = (const float*)d_in[19];
  const float* skipA  = (const float*)d_in[20];
  const float* skipB  = (const float*)d_in[21];
  const float* prAB   = (const float*)d_in[22];
  const float* prBA   = (const float*)d_in[23];
  const float* time_w = (const float*)d_in[24];
  const float* time_b = (const float*)d_in[25];
  const int*   idxAB  = (const int*)d_in[26];
  const int*   idxBA  = (const int*)d_in[27];

  uintptr_t p = (uintptr_t)d_ws;
  auto carve = [&](size_t bytes) -> void* {
    void* r = (void*)p;
    p += (bytes + 255) & ~(size_t)255;
    return r;
  };
  float*  kqvA   = (float*)carve((size_t)N_A * W3F * 4);    // q / k_src / v_src (A)
  float*  kqvB   = (float*)carve((size_t)N_B * W3F * 4);
  float*  WcA    = (float*)carve((size_t)IN_C * W3F * 4);   // composed f32 weights
  float*  bcA    = (float*)carve((size_t)W3F * 4);
  float*  WcB    = (float*)carve((size_t)IN_C * W3F * 4);
  float*  bcB    = (float*)carve((size_t)W3F * 4);
  bf16_t* pWcA   = (bf16_t*)carve((size_t)IN_C * W3F * 2);  // packed B fragments
  bf16_t* pWcB   = (bf16_t*)carve((size_t)IN_C * W3F * 2);
  bf16_t* pWedge = (bf16_t*)carve((size_t)E_PAD * D_DIM * 2);
  bf16_t* pWoutA = (bf16_t*)carve((size_t)256 * F_DIM * 2);
  bf16_t* pWoutB = (bf16_t*)carve((size_t)256 * F_DIM * 2);
  bf16_t* pXA    = (bf16_t*)carve((size_t)N_A * IN_C * 2);  // packed A fragments
  bf16_t* pXB    = (bf16_t*)carve((size_t)N_B * IN_C * 2);
  float*  EF     = (float*)carve((size_t)2 * NEDGE * 64 * 4);
  float*  alpha  = (float*)carve((size_t)2 * NEDGE * H_NUM * 4);
  float*  mbuf   = (float*)carve((size_t)N_TOT * H_NUM * 4);
  float*  sbuf   = (float*)carve((size_t)N_TOT * H_NUM * 4);
  float*  agg    = (float*)carve((size_t)N_TOT * 256 * 4);

  hipMemsetAsync(sbuf, 0, (size_t)N_TOT * H_NUM * 4, stream);
  hipMemsetAsync(agg,  0, (size_t)N_TOT * 256 * 4, stream);
  init_neg_kernel<<<(N_TOT * H_NUM + 255) / 256, 256, 0, stream>>>(mbuf, N_TOT * H_NUM);

  // weight prep: compose, then pack everything into fragment-major bf16
  compose_kernel<<<(2 * W3F * 129 + 255) / 256, 256, 0, stream>>>(
      WkqvA, bkqvA, WkqvB, bkqvB, Wk_rel, Wv_rel, WcA, bcA, WcB, bcB);
  pack_b_kernel<<<(IN_C * W3F / 16 + 255) / 256, 256, 0, stream>>>(WcA, pWcA, IN_C, W3F, IN_C);
  pack_b_kernel<<<(IN_C * W3F / 16 + 255) / 256, 256, 0, stream>>>(WcB, pWcB, IN_C, W3F, IN_C);
  pack_b_kernel<<<(E_PAD * D_DIM / 16 + 255) / 256, 256, 0, stream>>>(W_edge, pWedge, E_PAD, D_DIM, E_DIM);
  pack_b_kernel<<<(256 * F_DIM / 16 + 255) / 256, 256, 0, stream>>>(WoutA, pWoutA, 256, F_DIM, 256);
  pack_b_kernel<<<(256 * F_DIM / 16 + 255) / 256, 256, 0, stream>>>(WoutB, pWoutB, 256, F_DIM, 256);
  pack_a_kernel<<<((size_t)N_A * IN_C / 16 + 255) / 256, 256, 0, stream>>>(x_A, pXA, N_A, IN_C);
  pack_a_kernel<<<((size_t)N_B * IN_C / 16 + 255) / 256, 256, 0, stream>>>(x_B, pXB, N_B, IN_C);

  // fused kqv + relation transform GEMMs
  gemm_kqv_kernel<<<dim3(N_A / 16, W3F / 64), 32, 0, stream>>>(pXA, pWcA, bcA, kqvA);
  gemm_kqv_kernel<<<dim3(N_B / 16, W3F / 64), 32, 0, stream>>>(pXB, pWcB, bcB, kqvB);

  // edge features (time encoding + msg -> D) via LDS-staged WMMA
  edge_feat_kernel<<<2 * NEDGE / 16, 32, 0, stream>>>(
      idxAB, idxBA, luA, luB, tAB, tBA, msgAB, msgBA, time_w, time_b, pWedge, b_edge, EF);

  // segment-softmax attention + scatter aggregation
  alpha_kernel<<<(int)((2LL * NEDGE * 4 + 255) / 256), 256, 0, stream>>>(
      idxAB, idxBA, kqvA, kqvB, prAB, prBA, alpha, mbuf);
  softmax_kernel<<<(int)((2LL * NEDGE * H_NUM + 255) / 256), 256, 0, stream>>>(
      idxAB, idxBA, alpha, mbuf, sbuf);
  scatter_kernel<<<(int)((2LL * NEDGE * 16 + 255) / 256), 256, 0, stream>>>(
      idxAB, idxBA, kqvA, kqvB, EF, alpha, sbuf, agg);

  // gated output projections
  out_proj_kernel<<<dim3(N_A / 16, 2), 32, 0, stream>>>(
      agg, pWoutA, boutA, x_A, skipA, (float*)d_out);
  out_proj_kernel<<<dim3(N_B / 16, 2), 32, 0, stream>>>(
      agg + (size_t)N_A * 256, pWoutB, boutB, x_B, skipB,
      (float*)d_out + (size_t)N_A * F_DIM);
}
